// BRUnit_69836168233439
// MI455X (gfx1250) — matrix operand order repaired and checked by reading, not compiled, not run on hardware
//
#include <hip/hip_runtime.h>
#include <hip/hip_bf16.h>

// ---------------------------------------------------------------------------
// Types for CDNA5 WMMA (wave32). bf16 operands, f32 accumulate.
// ---------------------------------------------------------------------------
typedef __bf16 bf16_t;
typedef __attribute__((ext_vector_type(16))) __bf16 v16bf;
typedef __attribute__((ext_vector_type(8)))  __bf16 v8bf;
typedef __attribute__((ext_vector_type(8)))  float   v8f;
typedef int v4i __attribute__((__vector_size__(16)));

#define LSTM_H 1024
#define LSTM_B 128

static __device__ __forceinline__ v8f wmma_bf16(v16bf a, v16bf b, v8f c) {
  // D = A(16x32) * B(32x16) + C, f32 accum.
  return __builtin_amdgcn_wmma_f32_16x16x32_bf16(
      /*neg_a=*/false, a, /*neg_b=*/false, b,
      /*c_mod=*/(short)0, c, /*reuse_a=*/false, /*reuse_b=*/false);
}

// A-matrix 16x32 bf16 fragment (this code's contiguous k order; B packed alike):
// lanes 0-15: M=lane,    K = k0+0..15
// lanes16-31: M=lane-16, K = k0+16..31
static __device__ __forceinline__ v16bf load_a_frag(const bf16_t* base,
                                                    long rowStride, int lane) {
  int r  = lane & 15;
  int hi = lane >> 4;
  const bf16_t* p = base + (long)r * rowStride + (hi ? 16 : 0);
  v8bf lo = *(const v8bf*)(p);
  v8bf hh = *(const v8bf*)(p + 8);
  v16bf a;
#pragma unroll
  for (int i = 0; i < 8; ++i) { a[i] = lo[i]; a[i + 8] = hh[i]; }
  return a;
}

// B fragment from LDS, weights pre-packed in fragment order:
// element offset = ((chunk*4 + gate)*32 + lane) * 16  -> 32B per lane, linear.
static __device__ __forceinline__ v16bf lds_b_frag(const bf16_t* lds_base,
                                                   int chunk, int gate, int lane) {
  return *(const v16bf*)(lds_base + (((long)chunk * 4 + gate) * 32 + lane) * 16);
}

static __device__ __forceinline__ float sigmoidf_(float x) {
  return 1.0f / (1.0f + expf(-x));
}

// ---------------------------------------------------------------------------
// Global -> LDS bulk copy (16B units). Uses CDNA5 async-to-LDS path when the
// toolchain exposes the builtins; otherwise plain load+ds_store.
// ---------------------------------------------------------------------------
#if __has_builtin(__builtin_amdgcn_global_load_async_to_lds_b128) && \
    __has_builtin(__builtin_amdgcn_s_wait_asynccnt)
#define LSTM_ASYNC_LDS 1
#else
#define LSTM_ASYNC_LDS 0
#endif

static __device__ __forceinline__ void copy_g2l_16B(bf16_t* lds_dst,
                                                    const bf16_t* gsrc,
                                                    int units16, int tid) {
#if LSTM_ASYNC_LDS
  for (int i = tid; i < units16; i += 256) {
    __builtin_amdgcn_global_load_async_to_lds_b128(
        (__attribute__((address_space(1))) v4i*)(gsrc + (long)i * 8),
        (__attribute__((address_space(3))) v4i*)(lds_dst + (long)i * 8),
        0, 0);
  }
#else
  for (int i = tid; i < units16; i += 256) {
    ((uint4*)lds_dst)[i] = ((const uint4*)gsrc)[i];
  }
#endif
}

static __device__ __forceinline__ void copy_g2l_fence() {
#if LSTM_ASYNC_LDS
  __builtin_amdgcn_s_wait_asynccnt(0);
#endif
  __syncthreads();
}

// Grid-wide barrier: monotonic counter, generation-based.
static __device__ __forceinline__ void grid_sync(unsigned int* cnt,
                                                 unsigned int nblk,
                                                 unsigned int gen) {
  __threadfence();
  __syncthreads();
  if (threadIdx.x == 0) {
    atomicAdd(cnt, 1u);
    unsigned int target = gen * nblk;
    while (*(volatile unsigned int*)cnt < target) {
      __builtin_amdgcn_s_sleep(2);
    }
  }
  __syncthreads();
  __threadfence();
}

// ---------------------------------------------------------------------------
// Persistent LSTM layer. H=1024, B=128. Grid = 64 blocks x 256 threads.
// Block b owns hidden slice j0 = b*16 (all 4 gates for those units); its
// weight slices (64 rows x K, fragment-packed) live in LDS for the whole T
// loop. Wave w owns batch rows m0 = w*16; cell state c stays in VGPRs.
// ---------------------------------------------------------------------------
__global__ void __launch_bounds__(256)
lstm_layer_persistent(const bf16_t* __restrict__ Xseq, long xRowStride, long xTStride,
                      const bf16_t* __restrict__ Pih, int Kx,   // packed [blk][c][g][lane][16]
                      const bf16_t* __restrict__ Phh,           // packed, K = 1024
                      const float* __restrict__ b_ih, const float* __restrict__ b_hh,
                      bf16_t* __restrict__ Hout, int writeAll,
                      bf16_t* __restrict__ hbuf,                // [2, B, H] double buffer
                      unsigned int* __restrict__ sync_cnt,
                      int T) {
  // 256 KB static LDS: [0, 64K) packed Wih slice (Kx<=1024), [64K, 128K) Whh.
  __shared__ __align__(128) bf16_t lds_w[131072];
  bf16_t* lds_wih = lds_w;
  bf16_t* lds_whh = lds_w + 65536;

  const int tid  = threadIdx.x;
  const int lane = tid & 31;
  const int wave = tid >> 5;
  const int m0   = wave * 16;            // batch row base for this wave
  const int j0   = blockIdx.x * 16;      // hidden column base for this block
  const int n    = lane & 15;
  const int j    = j0 + n;               // this lane's hidden column (C/D layout)
  const int hi   = lane >> 4;
  const unsigned int nblk = gridDim.x;
  const int nkx = Kx / 32;               // k-chunks for input projection
  const int nkh = LSTM_H / 32;

  // ---- one-time: stage this block's packed weight slices into LDS ----
  copy_g2l_16B(lds_wih, Pih + (long)blockIdx.x * ((long)Kx * 64), Kx * 8, tid);
  copy_g2l_16B(lds_whh, Phh + (long)blockIdx.x * 65536, 8192, tid);
  copy_g2l_fence();

  // Combined biases for this lane's hidden unit (constant over T).
  const float bias_i = b_ih[0 * LSTM_H + j] + b_hh[0 * LSTM_H + j];
  const float bias_f = b_ih[1 * LSTM_H + j] + b_hh[1 * LSTM_H + j];
  const float bias_g = b_ih[2 * LSTM_H + j] + b_hh[2 * LSTM_H + j];
  const float bias_o = b_ih[3 * LSTM_H + j] + b_hh[3 * LSTM_H + j];

  v8f c_state = {};                      // cell state, persistent in VGPRs

  for (int t = 0; t < T; ++t) {
    const bf16_t* hprev = hbuf + (size_t)(t & 1) * LSTM_B * LSTM_H;
    bf16_t*       hnext = hbuf + (size_t)((t + 1) & 1) * LSTM_B * LSTM_H;

    v8f acc_i = {}, acc_f = {}, acc_g = {}, acc_o = {};

    // ---- input projection: x_t rows from global, weights from LDS -------
    const bf16_t* xt = Xseq + (long)t * xTStride + (long)m0 * xRowStride;
    if (t + 1 < T) __builtin_prefetch(xt + xTStride, 0, 1);  // next timestep
    for (int c = 0; c < nkx; ++c) {
      v16bf a = load_a_frag(xt + c * 32, xRowStride, lane);
      acc_i = wmma_bf16(a, lds_b_frag(lds_wih, c, 0, lane), acc_i);
      acc_f = wmma_bf16(a, lds_b_frag(lds_wih, c, 1, lane), acc_f);
      acc_g = wmma_bf16(a, lds_b_frag(lds_wih, c, 2, lane), acc_g);
      acc_o = wmma_bf16(a, lds_b_frag(lds_wih, c, 3, lane), acc_o);
    }

    // ---- recurrent projection: h_{t-1} rows from global, W from LDS -----
    const bf16_t* hrow = hprev + (long)m0 * LSTM_H;
    for (int c = 0; c < nkh; ++c) {
      v16bf a = load_a_frag(hrow + c * 32, LSTM_H, lane);
      acc_i = wmma_bf16(a, lds_b_frag(lds_whh, c, 0, lane), acc_i);
      acc_f = wmma_bf16(a, lds_b_frag(lds_whh, c, 1, lane), acc_f);
      acc_g = wmma_bf16(a, lds_b_frag(lds_whh, c, 2, lane), acc_g);
      acc_o = wmma_bf16(a, lds_b_frag(lds_whh, c, 3, lane), acc_o);
    }

    // ---- elementwise cell update (C/D layout: vgpr r -> row, lane -> col)
    v8f hval;
#pragma unroll
    for (int r = 0; r < 8; ++r) {
      float iv = sigmoidf_(acc_i[r] + bias_i);
      float fv = sigmoidf_(acc_f[r] + bias_f);
      float gv = tanhf(acc_g[r] + bias_g);
      float ov = sigmoidf_(acc_o[r] + bias_o);
      float cv = fv * c_state[r] + iv * gv;
      c_state[r] = cv;
      hval[r] = ov * tanhf(cv);
    }

    // ---- store h tile ---------------------------------------------------
    const int mbase = m0 + (hi ? 8 : 0);
#pragma unroll
    for (int r = 0; r < 8; ++r) {
      bf16_t hb = (bf16_t)hval[r];
      hnext[(long)(mbase + r) * LSTM_H + j] = hb;
      if (writeAll) {
        Hout[((long)t * LSTM_B + (mbase + r)) * LSTM_H + j] = hb;
      } else if (t == T - 1) {
        Hout[(long)(mbase + r) * LSTM_H + j] = hb;
      }
    }

    grid_sync(sync_cnt, nblk, (unsigned int)(t + 1));
  }
}

// ---------------------------------------------------------------------------
// Weight packing: f32 [4*H, Kx] -> bf16 fragment order
// P[v] with v = ((b*nk + c)*4 + g)*32 + l ; each v covers 16 bf16:
//   W[g*1024 + b*16 + (l&15)][c*32 + (l>>4)*16 .. +16)
// ---------------------------------------------------------------------------
__global__ void pack_w_kernel(const float* __restrict__ W,
                              bf16_t* __restrict__ P, int Kx) {
  long v = blockIdx.x * (long)blockDim.x + threadIdx.x;
  int nk = Kx / 32;
  long total = (long)64 * nk * 4 * 32;
  if (v >= total) return;
  int  l   = (int)(v & 31);
  int  g   = (int)((v >> 5) & 3);
  long rem = v >> 7;
  long c   = rem % nk;
  long b   = rem / nk;
  long row = (long)g * LSTM_H + b * 16 + (l & 15);
  long col = c * 32 + (l >> 4) * 16;
  const float* src = W + row * Kx + col;
  bf16_t* dst = P + v * 16;
#pragma unroll
  for (int i = 0; i < 16; ++i) dst[i] = (bf16_t)src[i];
}

// ---------------------------------------------------------------------------
// Final readout: out[b,r] = elu(h_last[b,:] . W_br[r,:] + b_br[r])
// ---------------------------------------------------------------------------
__global__ void __launch_bounds__(256)
br_elu_kernel(const bf16_t* __restrict__ hlast,
              const float* __restrict__ Wbr, const float* __restrict__ bbr,
              float* __restrict__ out) {
  int idx = blockIdx.x * blockDim.x + threadIdx.x;  // B*BR = 32768
  int b = idx >> 8;
  int r = idx & 255;
  const bf16_t* hp = hlast + (long)b * LSTM_H;
  const float*  wp = Wbr + (long)r * LSTM_H;
  float acc = bbr[r];
  for (int k = 0; k < LSTM_H; ++k) acc += (float)hp[k] * wp[k];
  out[idx] = acc > 0.0f ? acc : (expf(acc) - 1.0f);
}

// ---------------------------------------------------------------------------
// Helpers: f32 -> bf16 conversion, zero-fill.
// ---------------------------------------------------------------------------
__global__ void f32_to_bf16_kernel(const float* __restrict__ in,
                                   bf16_t* __restrict__ out, long n) {
  long i = blockIdx.x * (long)blockDim.x + threadIdx.x;
  if (i < n) out[i] = (bf16_t)in[i];
}

__global__ void zero_u32_kernel(unsigned int* __restrict__ p, long n) {
  long i = blockIdx.x * (long)blockDim.x + threadIdx.x;
  if (i < n) p[i] = 0u;
}

// ---------------------------------------------------------------------------
// Host launcher
// ---------------------------------------------------------------------------
extern "C" void kernel_launch(void* const* d_in, const int* in_sizes, int n_in,
                              void* d_out, int out_size, void* d_ws, size_t ws_size,
                              hipStream_t stream) {
  (void)in_sizes; (void)n_in; (void)out_size; (void)ws_size;
  const int B = 128, T = 256, D = 512, H = 1024;

  const float* X     = (const float*)d_in[0];
  const float* W_ih0 = (const float*)d_in[1];
  const float* W_hh0 = (const float*)d_in[2];
  const float* b_ih0 = (const float*)d_in[3];
  const float* b_hh0 = (const float*)d_in[4];
  const float* W_ih1 = (const float*)d_in[5];
  const float* W_hh1 = (const float*)d_in[6];
  const float* b_ih1 = (const float*)d_in[7];
  const float* b_hh1 = (const float*)d_in[8];
  const float* W_br  = (const float*)d_in[9];
  const float* b_br  = (const float*)d_in[10];
  float* out = (float*)d_out;

  // ---- workspace bump allocator (256B aligned) ----
  char* ws = (char*)d_ws;
  auto alloc = [&](size_t bytes) -> void* {
    void* p = (void*)ws;
    ws += (bytes + 255) & ~(size_t)255;
    return p;
  };
  const long nX    = (long)B * T * D;        // 16.8M
  const long nW512 = (long)4 * H * 512;      // 2.1M  (packed size == dense size)
  const long nW1k  = (long)4 * H * H;        // 4.2M
  const long nHseq = (long)T * B * H;        // 33.6M
  const long nHbuf = (long)2 * B * H;

  unsigned int* sync0 = (unsigned int*)alloc(256);
  unsigned int* sync1 = (unsigned int*)alloc(256);
  bf16_t* Xbf    = (bf16_t*)alloc(nX * 2);
  bf16_t* Pih0   = (bf16_t*)alloc(nW512 * 2);  // packed Wih0, Kx=512
  bf16_t* Phh0   = (bf16_t*)alloc(nW1k * 2);   // packed Whh0
  bf16_t* Pih1   = (bf16_t*)alloc(nW1k * 2);   // packed Wih1, Kx=1024
  bf16_t* Phh1   = (bf16_t*)alloc(nW1k * 2);   // packed Whh1
  bf16_t* Hseq0  = (bf16_t*)alloc(nHseq * 2);  // layer0 outputs, all t
  bf16_t* Hlast1 = (bf16_t*)alloc((long)B * H * 2);
  bf16_t* hbuf0  = (bf16_t*)alloc(nHbuf * 2);
  bf16_t* hbuf1  = (bf16_t*)alloc(nHbuf * 2);

  // ---- init: zero sync counters and h double buffers ----
  zero_u32_kernel<<<1, 128, 0, stream>>>(sync0, 128);  // covers both counters
  zero_u32_kernel<<<(nHbuf / 2 + 255) / 256, 256, 0, stream>>>((unsigned int*)hbuf0, nHbuf / 2);
  zero_u32_kernel<<<(nHbuf / 2 + 255) / 256, 256, 0, stream>>>((unsigned int*)hbuf1, nHbuf / 2);

  // ---- convert X, pack weights into WMMA fragment order ----
  f32_to_bf16_kernel<<<(int)((nX + 255) / 256), 256, 0, stream>>>(X, Xbf, nX);
  auto pack = [&](const float* src, bf16_t* dst, int Kx) {
    long thr = (long)64 * (Kx / 32) * 4 * 32;
    pack_w_kernel<<<(int)((thr + 255) / 256), 256, 0, stream>>>(src, dst, Kx);
  };
  pack(W_ih0, Pih0, D);
  pack(W_hh0, Phh0, H);
  pack(W_ih1, Pih1, H);
  pack(W_hh1, Phh1, H);

  // ---- layer 0: X[B,T,D] (rowStride=T*D, tStride=D), Kx=512 ----
  lstm_layer_persistent<<<H / 16, 256, 0, stream>>>(
      Xbf, (long)T * D, (long)D,
      Pih0, D, Phh0, b_ih0, b_hh0,
      Hseq0, /*writeAll=*/1, hbuf0, sync0, T);

  // ---- layer 1: Hseq0[T,B,H] (rowStride=H, tStride=B*H), Kx=1024 ----
  lstm_layer_persistent<<<H / 16, 256, 0, stream>>>(
      Hseq0, (long)H, (long)B * H,
      Pih1, H, Phh1, b_ih1, b_hh1,
      Hlast1, /*writeAll=*/0, hbuf1, sync1, T);

  // ---- readout: [128,1024] x [1024,256] + bias, ELU ----
  br_elu_kernel<<<(B * 256) / 256, 256, 0, stream>>>(Hlast1, W_br, b_br, out);
}
